// GraphSageNet_67860483277516
// MI455X (gfx1250) — matrix operand order repaired and checked by reading, not compiled, not run on hardware
//
#include <hip/hip_runtime.h>
#include <hip/hip_bf16.h>

typedef __attribute__((ext_vector_type(16))) _Float16 v16h;
typedef __attribute__((ext_vector_type(8)))  _Float16 v8h;
typedef __attribute__((ext_vector_type(8)))  float    v8f;

// 16-byte int vector matching the async-LDS builtin's parameter type
typedef int vsi4 __attribute__((vector_size(16)));
typedef __attribute__((address_space(1))) vsi4 g_vsi4;
typedef __attribute__((address_space(3))) vsi4 l_vsi4;

#define WMMA16x16x32(A, B, C) \
  __builtin_amdgcn_wmma_f32_16x16x32_f16(false, (A), false, (B), (short)0, (C), false, false)

#define CAT16(lo, hi_) __builtin_shufflevector((lo), (hi_), 0,1,2,3,4,5,6,7,8,9,10,11,12,13,14,15)

// Async global->LDS copy path (CDNA5, tracked by ASYNCcnt)
#if defined(__has_builtin)
#if __has_builtin(__builtin_amdgcn_global_load_async_to_lds_b128) && \
    __has_builtin(__builtin_amdgcn_s_wait_asynccnt)
#define HAVE_ASYNC_LDS 1
#endif
#endif
#ifndef HAVE_ASYNC_LDS
#define HAVE_ASYNC_LDS 0
#endif

// ---------------------------------------------------------------------------
// Problem constants
// ---------------------------------------------------------------------------
#define BB   32
#define NN   32
#define HID  256
#define OUTD 128
#define C1O  64      // conv1 out channels
#define C2O  128     // conv2 out channels
#define PIX  1024    // 32x32
#define K2   576     // conv2 GEMM K = 9*64

// Workspace layout (bytes)
#define WS_W2H   0                          // f16 [128][576]       147456 B
#define WS_HBUF  147456                     // f16 [1024][256]      524288 B
#define WS_M1T   (147456 + 524288)          // f16 [256][256] (n-major) 131072 B
#define WS_M2T   (147456 + 524288 + 131072) // f16 [128][256] (n-major)  65536 B

// ---------------------------------------------------------------------------
// Kernel 1: weight prep (f32 -> f16, permutes / SAGE algebra folding)
// M1T[n][k] = s1_rw[n][k] - s1_lw[n][k]/31   (transposed so WMMA B fragments
// M2T[n][k] = s2_rw[n][k] - s2_lw[n][k]/31    are contiguous 32 B loads)
// ---------------------------------------------------------------------------
__global__ __launch_bounds__(256)
void gsage_prep_kernel(const float* __restrict__ conv2_w,
                       const float* __restrict__ s1_lw, const float* __restrict__ s1_rw,
                       const float* __restrict__ s2_lw, const float* __restrict__ s2_rw,
                       _Float16* __restrict__ W2h, _Float16* __restrict__ M1T,
                       _Float16* __restrict__ M2T) {
  const int i = blockIdx.x * 256 + threadIdx.x;
  if (i < C2O * K2) {
    // W2h[o][khw*64 + ci] = conv2_w[o][ci][kh][kw]
    const int o = i / K2, r = i % K2, khw = r >> 6, ci = r & 63;
    W2h[i] = (_Float16)conv2_w[o * K2 + ci * 9 + khw];
  } else if (i < C2O * K2 + HID * HID) {
    const int j = i - C2O * K2;               // j = n*256 + k
    M1T[j] = (_Float16)(s1_rw[j] - s1_lw[j] * (1.0f / 31.0f));
  } else if (i < C2O * K2 + HID * HID + OUTD * HID) {
    const int j = i - C2O * K2 - HID * HID;   // j = n*256 + k
    M2T[j] = (_Float16)(s2_rw[j] - s2_lw[j] * (1.0f / 31.0f));
  }
}

// ---------------------------------------------------------------------------
// Kernel 2: fused conv1 -> conv2(WMMA implicit GEMM) -> avgpool -> fc -> mask
// One workgroup (256 threads / 8 waves) per image. Dynamic LDS (316736 B,
// fits CDNA5's 320 KB/WGP):
//   xs     f32 [3][34][34]     13872 B  @ 0
//   w1s    f32 [64][27]         6912 B  @ 13872
//   pooled f32 [128]             512 B  @ 20784
//   z1     f16 [34*34][64]    147968 B  @ 21312   (pixel-major, channel inner)
//   W2s    f16 [128][576]     147456 B  @ 169280  (conv2 weights, LDS-resident)
// The W2s fill is issued async (ASYNCcnt) and hidden under the conv1 phase.
// ---------------------------------------------------------------------------
#define SM_XS   0
#define SM_W1S  13872
#define SM_POOL 20784
#define SM_Z1   21312
#define SM_W2S  169280
#define SM_TOT  316736

__global__ __launch_bounds__(256)
void gsage_conv_kernel(const float* __restrict__ x, const float* __restrict__ mask,
                       const float* __restrict__ w1, const float* __restrict__ b1,
                       const _Float16* __restrict__ W2h, const float* __restrict__ b2,
                       const float* __restrict__ fcw, const float* __restrict__ fcb,
                       _Float16* __restrict__ Hbuf) {
  extern __shared__ __align__(32) char smem[];
  float*    xs     = (float*)(smem + SM_XS);
  float*    w1s    = (float*)(smem + SM_W1S);
  float*    pooled = (float*)(smem + SM_POOL);
  _Float16* z1     = (_Float16*)(smem + SM_Z1);
  _Float16* W2s    = (_Float16*)(smem + SM_W2S);

  const int tid = threadIdx.x;
  const int img = blockIdx.x;           // 0..1023 = b*32 + node

  // ---- phase 1: stage conv2 weights (async), input, conv1 weights ----
#if HAVE_ASYNC_LDS
  {
    g_vsi4* gsrc = (g_vsi4*)W2h;
    l_vsi4* ldst = (l_vsi4*)(smem + SM_W2S);
    // 147456 B / 16 B per lane / 256 threads = 36 async b128 per thread
    for (int k = tid; k < (C2O * K2 * 2) / 16; k += 256)
      __builtin_amdgcn_global_load_async_to_lds_b128(gsrc + k, ldst + k, 0, 0);
  }
#else
  {
    const int4* src = (const int4*)W2h;
    int4* dst = (int4*)W2s;
    for (int i = tid; i < (C2O * K2 * 2) / 16; i += 256) dst[i] = src[i];
  }
#endif
  for (int i = tid; i < 3 * 34 * 34; i += 256) {
    const int ci = i / 1156, r = i % 1156, py = r / 34, px = r % 34;
    float v = 0.0f;
    if (py >= 1 && py <= 32 && px >= 1 && px <= 32)
      v = x[(size_t)img * 3072 + ci * PIX + (py - 1) * 32 + (px - 1)];
    xs[i] = v;
  }
  for (int i = tid; i < C1O * 27; i += 256) w1s[i] = w1[i];
  if (tid < C2O) pooled[tid] = 0.0f;
  {
    unsigned int* z32 = (unsigned int*)z1;
    for (int i = tid; i < (1156 * C1O) / 2; i += 256) z32[i] = 0u;
  }
  __syncthreads();

  // ---- phase 2: conv1 + relu -> z1 (padded, [pix][ch] f16) ----
  // (the async W2s fill proceeds in the background during this phase)
  for (int idx = tid; idx < C1O * PIX; idx += 256) {
    const int ch = idx >> 10;
    const int p  = idx & 1023;
    const int y = p >> 5, xp = p & 31;
    float acc = b1[ch];
    const float* wr = &w1s[ch * 27];
#pragma unroll
    for (int ci = 0; ci < 3; ++ci)
#pragma unroll
      for (int kh = 0; kh < 3; ++kh)
#pragma unroll
        for (int kw = 0; kw < 3; ++kw)
          acc += xs[ci * 1156 + (y + kh) * 34 + (xp + kw)] * wr[ci * 9 + kh * 3 + kw];
    acc = fmaxf(acc, 0.0f);
    z1[((y + 1) * 34 + (xp + 1)) * C1O + ch] = (_Float16)acc;
  }
#if HAVE_ASYNC_LDS
  __builtin_amdgcn_s_wait_asynccnt(0);   // W2s fill complete for this wave
#endif
  __syncthreads();

  // ---- phase 3: conv2 implicit GEMM via WMMA. M=128, N=1024 px, K=576 ----
  // Register blocking: 4 M-tiles x 2 N-tiles per wave pass. A and B both LDS.
  const int wv   = tid >> 5;
  const int lane = tid & 31;
  const int hi   = (lane >= 16) ? 1 : 0;
  const int col  = lane & 15;
  const int mblock = wv & 1;   // mtiles [4*mblock .. 4*mblock+3]
  const int wcol   = wv >> 1;  // 0..3: owns 16 n-tiles

#pragma unroll 1
  for (int jp = 0; jp < 8; ++jp) {
    const int n0 = wcol * 16 + jp * 2;     // n-tile pair (n0, n0+1)
    const int p0 = n0 * 16 + col;
    const int p1 = p0 + 16;
    const int y0 = p0 >> 5, x0 = p0 & 31;
    const int y1 = p1 >> 5, x1 = p1 & 31;

    v8f acc0[4] = {v8f{}, v8f{}, v8f{}, v8f{}};
    v8f acc1[4] = {v8f{}, v8f{}, v8f{}, v8f{}};

#pragma unroll 1
    for (int ks = 0; ks < 18; ++ks) {
      const int khw = ks >> 1;
      const int kh = khw / 3, kw = khw % 3;
      const int cioff = (ks & 1) * 32 + hi * 16;
      // B fragments: one pixel, 16 contiguous channels -> single aligned v16h
      const v16h b0 = *(const v16h*)&z1[((y0 + kh) * 34 + (x0 + kw)) * C1O + cioff];
      const v16h b1 = *(const v16h*)&z1[((y1 + kh) * 34 + (x1 + kw)) * C1O + cioff];
#pragma unroll
      for (int ai = 0; ai < 4; ++ai) {
        const int M = (mblock * 4 + ai) * 16 + col;
        const int base = M * K2 + ks * 32 + hi * 16;
        const v8h alo = *(const v8h*)&W2s[base];
        const v8h ahi = *(const v8h*)&W2s[base + 8];
        const v16h afrag = CAT16(alo, ahi);
        acc0[ai] = WMMA16x16x32(afrag, b0, acc0[ai]);
        acc1[ai] = WMMA16x16x32(afrag, b1, acc1[ai]);
      }
    }
    // fused bias + relu + global-average-pool accumulation (both N-tiles)
#pragma unroll
    for (int ai = 0; ai < 4; ++ai)
#pragma unroll
      for (int r = 0; r < 8; ++r) {
        const int ch = (mblock * 4 + ai) * 16 + r + hi * 8;
        const float bb = b2[ch];
        const float v = fmaxf(acc0[ai][r] + bb, 0.0f) + fmaxf(acc1[ai][r] + bb, 0.0f);
        atomicAdd(&pooled[ch], v);
      }
  }
  __syncthreads();

  // ---- phase 4: fc (128 -> 256) + mask, store f16 node embedding ----
  {
    float a = fcb[tid];
    float s = 0.0f;
    for (int k = 0; k < C2O; ++k) s += pooled[k] * fcw[tid * C2O + k];
    a += s * (1.0f / (float)PIX);        // mean pool folded into fc
    a *= mask[img];
    Hbuf[(size_t)img * HID + tid] = (_Float16)a;
  }
}

// ---------------------------------------------------------------------------
// Kernel 3: two SAGE layers per batch via WMMA.
// out_i = h_i @ M + (S@lw^T)/31 + lb, M precomputed f16 (n-major); relu after
// layer 1. One workgroup (256 threads / 8 waves) per batch.
// ---------------------------------------------------------------------------
__global__ __launch_bounds__(256)
void gsage_sage_kernel(const _Float16* __restrict__ Hbuf,
                       const _Float16* __restrict__ M1T, const _Float16* __restrict__ M2T,
                       const float* __restrict__ s1_lw, const float* __restrict__ s1_lb,
                       const float* __restrict__ s2_lw, const float* __restrict__ s2_lb,
                       float* __restrict__ out) {
  __shared__ __align__(32) _Float16 hA[NN * HID];   // 16 KB
  __shared__ __align__(32) _Float16 hB[NN * HID];   // 16 KB
  __shared__ float Sv[HID];
  __shared__ float Cv[HID];

  const int tid = threadIdx.x;
  const int b   = blockIdx.x;
  const int wv = tid >> 5, lane = tid & 31;
  const int hi = (lane >= 16) ? 1 : 0;
  const int col = lane & 15;

  for (int i = tid; i < NN * HID; i += 256) hA[i] = Hbuf[(size_t)b * NN * HID + i];
  __syncthreads();

  // S = sum_i h_i ; c1[o] = lb[o] + (S @ lw^T)[o]/31
  { float s = 0.0f;
    for (int i = 0; i < NN; ++i) s += (float)hA[i * HID + tid];
    Sv[tid] = s; }
  __syncthreads();
  { float a = 0.0f;
    for (int k = 0; k < HID; ++k) a += Sv[k] * s1_lw[tid * HID + k];
    Cv[tid] = s1_lb[tid] + a * (1.0f / 31.0f); }
  __syncthreads();

  // ---- SAGE1 GEMM: [32x256] @ [256x256], 32 tiles over 8 waves ----
#pragma unroll 1
  for (int t8 = 0; t8 < 4; ++t8) {
    const int T = wv + t8 * 8;
    const int mtile = T & 1, ntile = T >> 1;
    const int ncol = ntile * 16 + col;
    v8f acc = {};
#pragma unroll 1
    for (int ks = 0; ks < 8; ++ks) {
      const int Mrow = mtile * 16 + col;
      const int base = Mrow * HID + ks * 32 + hi * 16;
      const v8h alo = *(const v8h*)&hA[base];
      const v8h ahi = *(const v8h*)&hA[base + 8];
      const v16h afrag = CAT16(alo, ahi);
      // B fragment: 16 contiguous k for this lane's column (n-major layout)
      const v16h bfrag = *(const v16h*)&M1T[(size_t)ncol * HID + ks * 32 + hi * 16];
      acc = WMMA16x16x32(afrag, bfrag, acc);
    }
#pragma unroll
    for (int r = 0; r < 8; ++r) {
      const int node = mtile * 16 + r + hi * 8;
      hB[node * HID + ncol] = (_Float16)fmaxf(acc[r] + Cv[ncol], 0.0f);
    }
  }
  __syncthreads();

  // S2 / c2 over layer-1 output
  { float s = 0.0f;
    for (int i = 0; i < NN; ++i) s += (float)hB[i * HID + tid];
    Sv[tid] = s; }
  __syncthreads();
  if (tid < OUTD) {
    float a = 0.0f;
    for (int k = 0; k < HID; ++k) a += Sv[k] * s2_lw[tid * HID + k];
    Cv[tid] = s2_lb[tid] + a * (1.0f / 31.0f);
  }
  __syncthreads();

  // ---- SAGE2 GEMM: [32x256] @ [256x128], 16 tiles over 8 waves ----
#pragma unroll 1
  for (int t8 = 0; t8 < 2; ++t8) {
    const int T = wv + t8 * 8;
    const int mtile = T & 1, ntile = T >> 1;   // ntile 0..7
    const int ncol = ntile * 16 + col;
    v8f acc = {};
#pragma unroll 1
    for (int ks = 0; ks < 8; ++ks) {
      const int Mrow = mtile * 16 + col;
      const int base = Mrow * HID + ks * 32 + hi * 16;
      const v8h alo = *(const v8h*)&hB[base];
      const v8h ahi = *(const v8h*)&hB[base + 8];
      const v16h afrag = CAT16(alo, ahi);
      const v16h bfrag = *(const v16h*)&M2T[(size_t)ncol * HID + ks * 32 + hi * 16];
      acc = WMMA16x16x32(afrag, bfrag, acc);
    }
#pragma unroll
    for (int r = 0; r < 8; ++r) {
      const int node = mtile * 16 + r + hi * 8;
      out[((size_t)b * NN + node) * OUTD + ncol] = acc[r] + Cv[ncol];
    }
  }
}

// ---------------------------------------------------------------------------
// Host launcher
// ---------------------------------------------------------------------------
extern "C" void kernel_launch(void* const* d_in, const int* in_sizes, int n_in,
                              void* d_out, int out_size, void* d_ws, size_t ws_size,
                              hipStream_t stream) {
  const float* x       = (const float*)d_in[0];
  const float* mask    = (const float*)d_in[1];
  const float* conv1_w = (const float*)d_in[2];
  const float* conv1_b = (const float*)d_in[3];
  const float* conv2_w = (const float*)d_in[4];
  const float* conv2_b = (const float*)d_in[5];
  const float* fc_w    = (const float*)d_in[6];
  const float* fc_b    = (const float*)d_in[7];
  const float* s1_lw   = (const float*)d_in[8];
  const float* s1_lb   = (const float*)d_in[9];
  const float* s1_rw   = (const float*)d_in[10];
  const float* s2_lw   = (const float*)d_in[11];
  const float* s2_lb   = (const float*)d_in[12];
  const float* s2_rw   = (const float*)d_in[13];

  char* ws = (char*)d_ws;
  _Float16* W2h  = (_Float16*)(ws + WS_W2H);
  _Float16* Hbuf = (_Float16*)(ws + WS_HBUF);
  _Float16* M1T  = (_Float16*)(ws + WS_M1T);
  _Float16* M2T  = (_Float16*)(ws + WS_M2T);
  float* out = (float*)d_out;

  const int prep_elems = C2O * K2 + HID * HID + OUTD * HID;
  gsage_prep_kernel<<<(prep_elems + 255) / 256, 256, 0, stream>>>(
      conv2_w, s1_lw, s1_rw, s2_lw, s2_rw, W2h, M1T, M2T);

  gsage_conv_kernel<<<BB * NN, 256, SM_TOT, stream>>>(
      x, mask, conv1_w, conv1_b, W2h, conv2_b, fc_w, fc_b, Hbuf);

  gsage_sage_kernel<<<BB, 256, 0, stream>>>(
      Hbuf, M1T, M2T, s1_lw, s1_lb, s2_lw, s2_lb, out);
}
